// AChebyKANLinear_35914516529679
// MI455X (gfx1250) — matrix operand order repaired and checked by reading, not compiled, not run on hardware
//
#include <hip/hip_runtime.h>

typedef __attribute__((ext_vector_type(16))) _Float16 v16h;
typedef __attribute__((ext_vector_type(8)))  _Float16 v8h;
typedef __attribute__((ext_vector_type(8)))  float    v8f;

#define BATCH      32768
#define IN_DIM     256
#define OUT_DIM    256
#define NDEG       17               // Chebyshev degrees 0..16
#define DPC        18               // 17 cheby + 1 silu value per input channel
#define KTOT       (IN_DIM * DPC)   // 4608
#define KSTEPS     (KTOT / 32)      // 144 WMMA K-steps total
#define CHUNK_CH   16               // input channels staged per LDS chunk
#define CHUNK_K    (CHUNK_CH * DPC) // 288 K-values per chunk
#define CHUNK_KS   (CHUNK_K / 32)   // 9 K-steps per chunk
#define NCHUNKS    (IN_DIM / CHUNK_CH) // 16
#define M_TILE     64
#define LDS_STRIDE 304              // halves per A row (288 + pad), 32B-aligned rows
#define NTILES_N   (OUT_DIM / 16)   // 16

// ---------------------------------------------------------------------------
// Kernel 1: effective per-degree gate weights w[0..16].
// ---------------------------------------------------------------------------
__global__ void prep_gate_kernel(const float* __restrict__ logits,
                                 const float* __restrict__ gating_weights,
                                 float* __restrict__ gate) {
    int d = threadIdx.x;
    if (d >= NDEG) return;
    float w = gating_weights[d];
    if (d >= 9) {
        int e = d - 9;
        float le = logits[e];
        int rank = 0;
        for (int j = 0; j < 8; ++j) {
            float lj = logits[j];
            if (lj > le || (lj == le && j < e)) rank++;   // stable top_k tie-break
        }
        w = (rank < 4) ? (1.0f / (1.0f + __expf(-le))) : 0.0f;
    }
    gate[d] = w;
}

// ---------------------------------------------------------------------------
// Kernel 2: pack W (K=4608 x N=256) as f16 in WMMA B-fragment order.
//   n = ntile*16 + (lane & 15),  k = kstep*32 + (lane>>4)*16 + h
//   row k = i*18 + d : d<17 -> cheby[i][n][d]*gate[d] ; d==17 -> base_weight[n][i]
// ---------------------------------------------------------------------------
__global__ void pack_w_kernel(const float* __restrict__ cheby,
                              const float* __restrict__ basew,
                              const float* __restrict__ gate,
                              _Float16* __restrict__ Wp) {
    int idx = blockIdx.x * blockDim.x + threadIdx.x;   // 16*144*32 = 73728
    int lane  = idx & 31;
    int kstep = (idx >> 5) % KSTEPS;
    int ntile = idx / (32 * KSTEPS);
    int n     = ntile * 16 + (lane & 15);
    int kbase = kstep * 32 + ((lane >> 4) << 4);
    _Float16* dst = Wp + (size_t)idx * 16;
#pragma unroll
    for (int h = 0; h < 16; ++h) {
        int k = kbase + h;
        int i = k / DPC;
        int d = k - i * DPC;
        float v;
        if (d < NDEG) {
            v = cheby[((size_t)i * OUT_DIM + n) * NDEG + d] * gate[d];
        } else {
            v = basew[(size_t)n * IN_DIM + i];
        }
        dst[h] = (_Float16)v;
    }
}

// ---------------------------------------------------------------------------
// Kernel 3: fused activation + GEMM, software-pipelined.
// Block: 256 threads = 8 waves (4M x 2N); M_TILE=64, N=256 full.
// Loop body: issue async x(c+2), UNCONDITIONAL consume(c) (in-place WMMA
// accumulation), produce A(c+1) (VALU co-executes with XDL WMMA), wait+barrier.
// ---------------------------------------------------------------------------
__global__ __launch_bounds__(256, 1)
void cheby_gemm_kernel(const float* __restrict__ x,
                       const _Float16* __restrict__ Wp,
                       float* __restrict__ out) {
    __shared__ _Float16 smA[2][M_TILE * LDS_STRIDE];   // 2 x 38,912 B
    __shared__ float    smX[2][M_TILE * CHUNK_CH];     // 2 x 4,096  B

    const int tid   = threadIdx.x;
    const int lane  = tid & 31;
    const int wave  = tid >> 5;
    const int waveM = wave & 3;        // M sub-tile: rows waveM*16 .. +15
    const int waveN = wave >> 2;       // N half: cols waveN*128 .. +127
    const int blockM = blockIdx.x * M_TILE;

    v8f acc[8] = {};                   // 8 N-tiles of 16x16 f32

    // x async-stage mapping: one b128 (4 floats) per thread per chunk
    const int xr  = tid >> 2;          // row 0..63
    const int xc4 = (tid & 3) * 4;     // channel 0,4,8,12

    // activation producer mapping: thread -> (channel-in-chunk, 4 rows)
    const int ach = tid & 15;          // channel within chunk
    const int am0 = tid >> 4;          // rows am0, +16, +32, +48

    // WMMA A-fragment addressing (contiguous K, same order as packed B:
    //   lanes 0-15: K 0-15 ; lanes 16-31: K 16-31)
    const int arow = waveM * 16 + (lane & 15);
    const int sel  = (lane >> 4);

    // ---- helpers ---------------------------------------------------------
    auto issue_x = [&](int cc) {   // async-stage x(cc) into smX[cc&1]
        const float* gsrc =
            &x[(size_t)(blockM + xr) * IN_DIM + cc * CHUNK_CH + xc4];
        unsigned ldst = (unsigned)(unsigned long long)
            &smX[cc & 1][xr * CHUNK_CH + xc4];
        asm volatile("global_load_async_to_lds_b128 %0, %1, off"
                     :: "v"(ldst), "v"((unsigned long long)gsrc) : "memory");
    };

    auto produce = [&](int cc) {   // activations for chunk cc -> smA[cc&1]
        _Float16* Aw = smA[cc & 1];
        const float* Xs = smX[cc & 1];
#pragma unroll
        for (int r = 0; r < 4; ++r) {
            const int m = am0 + r * 16;
            const float xv = Xs[m * CHUNK_CH + ach];

            float u = tanhf(xv);
            u = fminf(fmaxf(u, -1.0f + 1e-6f), 1.0f - 1e-6f);
            float T[DPC];
            T[0] = 1.0f;
            T[1] = u;
            const float u2 = 2.0f * u;
#pragma unroll
            for (int d = 2; d < NDEG; ++d)
                T[d] = __builtin_fmaf(u2, T[d - 1], -T[d - 2]);
            T[17] = xv / (1.0f + __expf(-xv));   // silu row -> base_weight

            unsigned int* dst = (unsigned int*)&Aw[m * LDS_STRIDE + ach * DPC];
#pragma unroll
            for (int p = 0; p < DPC / 2; ++p) {
                union { _Float16 h[2]; unsigned int u32; } pk;
                pk.h[0] = (_Float16)T[2 * p];
                pk.h[1] = (_Float16)T[2 * p + 1];
                dst[p] = pk.u32;
            }
        }
    };

    auto consume = [&](int cc) {   // 9 K-steps x 8 N-tiles of WMMA
        const _Float16* Ab = smA[cc & 1];
#pragma unroll
        for (int ks = 0; ks < CHUNK_KS; ++ks) {
            const _Float16* ap = &Ab[arow * LDS_STRIDE + ks * 32];
            v8h alo = *(const v8h*)(ap + sel * 16);       // K sel*16 .. +7
            v8h ahi = *(const v8h*)(ap + sel * 16 + 8);   // K sel*16+8 .. +15
            v16h a = __builtin_shufflevector(alo, ahi,
                        0, 1, 2, 3, 4, 5, 6, 7, 8, 9, 10, 11, 12, 13, 14, 15);
            const int kstep_g = cc * CHUNK_KS + ks;
#pragma unroll
            for (int nt = 0; nt < 8; ++nt) {
                const int ntile = waveN * 8 + nt;
                const v16h b = *(const v16h*)&Wp[
                    ((((size_t)ntile * KSTEPS + kstep_g) * 32) + lane) * 16];
                acc[nt] = __builtin_amdgcn_wmma_f32_16x16x32_f16(
                    false, a, false, b, (short)0, acc[nt], false, false);
            }
        }
    };

    // ---- prologue: stage x(0), produce A(0), stage x(1) ------------------
    issue_x(0);
    asm volatile("s_wait_asynccnt 0x0" ::: "memory");
    __syncthreads();                       // x(0) ready for all waves

    produce(0);
    issue_x(1);
    asm volatile("s_wait_asynccnt 0x0" ::: "memory");
    __syncthreads();                       // A(0) + x(1) ready

    // ---- steady state ----------------------------------------------------
    for (int c = 0; c < NCHUNKS - 1; ++c) {
        if (c + 2 < NCHUNKS) issue_x(c + 2);   // uniform guard, acc untouched
        consume(c);                             // unconditional, in-place acc
        produce(c + 1);                         // overlaps with WMMA stream
        asm volatile("s_wait_asynccnt 0x0" ::: "memory");
        __syncthreads();
    }

    // ---- epilogue --------------------------------------------------------
    consume(NCHUNKS - 1);

    // ---- store C (nontemporal: output never re-read, keep L2 for W) ------
    const int nlo = lane & 15;
    const int mhi = (lane >> 4) * 8;
#pragma unroll
    for (int nt = 0; nt < 8; ++nt) {
        const int col = waveN * 128 + nt * 16 + nlo;
#pragma unroll
        for (int v = 0; v < 8; ++v) {
            const int row = blockM + waveM * 16 + mhi + v;
            __builtin_nontemporal_store(acc[nt][v], &out[(size_t)row * OUT_DIM + col]);
        }
    }
}

// ---------------------------------------------------------------------------
// Launcher. Inputs: 0=x, 1=t(unused), 2=logits, 3=cheby_coeffs, 4=base_weight,
//                   5=gating_weights, 6=arange(unused)
// Workspace: [0,256): gate[17] f32; [256, 256+2.36MB): packed W f16.
// ---------------------------------------------------------------------------
extern "C" void kernel_launch(void* const* d_in, const int* in_sizes, int n_in,
                              void* d_out, int out_size, void* d_ws, size_t ws_size,
                              hipStream_t stream) {
    const float* x      = (const float*)d_in[0];
    const float* logits = (const float*)d_in[2];
    const float* cheby  = (const float*)d_in[3];
    const float* basew  = (const float*)d_in[4];
    const float* gw     = (const float*)d_in[5];

    float*    gate = (float*)d_ws;
    _Float16* Wp   = (_Float16*)((char*)d_ws + 256);

    prep_gate_kernel<<<1, 32, 0, stream>>>(logits, gw, gate);

    pack_w_kernel<<<(NTILES_N * KSTEPS * 32) / 256, 256, 0, stream>>>(
        cheby, basew, gate, Wp);

    cheby_gemm_kernel<<<BATCH / M_TILE, 256, 0, stream>>>(
        x, Wp, (float*)d_out);
}
